// LSK_input_gate_dcnv3_87316685128225
// MI455X (gfx1250) — matrix operand order repaired and checked by reading, not compiled, not run on hardware
//
#include <hip/hip_runtime.h>
#include <hip/hip_bf16.h>

typedef __attribute__((ext_vector_type(16))) _Float16 v16h;
typedef __attribute__((ext_vector_type(8)))  _Float16 v8h;
typedef __attribute__((ext_vector_type(8)))  float    v8f;

#define LROWS 8192      // N*H*W = 2*64*64
#define CDIM  256
#define HWSZ  4096
#define KT8   8         // 256 / 32

// ---------------------------------------------------------------------------
// NCHW -> NHWC transpose, f32 + f16 copies
// ---------------------------------------------------------------------------
__global__ __launch_bounds__(256) void nchw_to_nhwc(const float* __restrict__ X,
                                                    float* __restrict__ Xf,
                                                    _Float16* __restrict__ Xh) {
  size_t idx = (size_t)blockIdx.x * 256 + threadIdx.x;   // over N*C*HW = 2M
  int hw = (int)(idx & (HWSZ - 1));
  int nc = (int)(idx >> 12);
  int c  = nc & (CDIM - 1);
  int n  = nc >> 8;
  float v = X[idx];
  size_t o = ((size_t)(n * HWSZ + hw)) * CDIM + c;
  Xf[o] = v;
  Xh[o] = (_Float16)v;
}

// ---------------------------------------------------------------------------
// Pack f32 weight [K=256, N] into wave32 WMMA B-fragment tiles (f16).
// Layout: h = ((nt*8 + kt)*32 + lane)*16 + e
//   lane<16 : column n = nt*16+lane,    element e -> k = kt*32 + e
//   lane>=16: column n = nt*16+lane-16, element e -> k = kt*32 + 16 + e
// Columns n >= N are zero-filled (NT may be padded to an even tile count).
// ---------------------------------------------------------------------------
__global__ __launch_bounds__(256) void pack_w(const float* __restrict__ W,
                                              _Float16* __restrict__ Wp,
                                              int N, int NT) {
  int total = NT * KT8 * 512;
  int h = blockIdx.x * 256 + threadIdx.x;
  if (h >= total) return;
  int e    = h & 15;
  int lane = (h >> 4) & 31;
  int kt   = (h >> 9) & (KT8 - 1);
  int nt   = h >> 12;                 // /(512*8)
  int n = nt * 16 + (lane & 15);
  int k = kt * 32 + ((lane >> 4) << 4) + e;
  float v = (n < N) ? W[(size_t)k * N + n] : 0.f;
  Wp[h] = (_Float16)v;
}

// ---------------------------------------------------------------------------
// GEMM: out[L,N] = A(f16,[L,256]) x Wp(packed f16) + bias
// One wave computes a 32x32 tile: 2 M-tiles x 2 N-tiles, 4 accumulators,
// K=256 fully unrolled -> 32 WMMA ops, A/B fragments reused 2x each.
// mode 0: store f32; mode 1: store f32 + f16; mode 2: gated NCHW store
// ---------------------------------------------------------------------------
__global__ __launch_bounds__(256) void gemm_wmma(const _Float16* __restrict__ A,
                                                 const _Float16* __restrict__ Wp,
                                                 const float* __restrict__ bias,
                                                 int N, int mode,
                                                 float* __restrict__ outF,
                                                 _Float16* __restrict__ outH,
                                                 const float* __restrict__ gateX,
                                                 float* __restrict__ outGate) {
  int wave = blockIdx.x * 8 + (threadIdx.x >> 5);
  int lane = threadIdx.x & 31;
  const int MT2 = LROWS / 32;           // 256 row-pair tiles
  int mt2 = wave & (MT2 - 1);
  int np  = wave / MT2;                 // n-tile pair index

  int r0 = mt2 << 5;
  int c0 = np << 5;
  int lane15 = lane & 15;
  int hiLane = lane >> 4;               // 0 or 1
  int ka     = hiLane << 4;             // 0 or 16

  // A fragments: lane<16 -> row +lane15, K {0..15}; lane>=16 -> K {16..31}
  const v8h* ap0 = (const v8h*)(A + (size_t)(r0 + lane15) * CDIM + ka);
  const v8h* ap1 = (const v8h*)(A + (size_t)(r0 + 16 + lane15) * CDIM + ka);
  const _Float16* wp0 = Wp + (size_t)(np * 2) * (KT8 * 512);
  const _Float16* wp1 = wp0 + (KT8 * 512);

  v8f acc00 = {}, acc01 = {}, acc10 = {}, acc11 = {};
#pragma unroll
  for (int kt = 0; kt < KT8; ++kt) {
    v16h a0 = __builtin_shufflevector(ap0[kt * 4], ap0[kt * 4 + 1],
                                      0,1,2,3,4,5,6,7,8,9,10,11,12,13,14,15);
    v16h a1 = __builtin_shufflevector(ap1[kt * 4], ap1[kt * 4 + 1],
                                      0,1,2,3,4,5,6,7,8,9,10,11,12,13,14,15);
    v16h b0 = *(const v16h*)(wp0 + (size_t)kt * 512 + lane * 16);
    v16h b1 = *(const v16h*)(wp1 + (size_t)kt * 512 + lane * 16);
    acc00 = __builtin_amdgcn_wmma_f32_16x16x32_f16(false, a0, false, b0, (short)0, acc00, false, false);
    acc01 = __builtin_amdgcn_wmma_f32_16x16x32_f16(false, a0, false, b1, (short)0, acc01, false, false);
    acc10 = __builtin_amdgcn_wmma_f32_16x16x32_f16(false, a1, false, b0, (short)0, acc10, false, false);
    acc11 = __builtin_amdgcn_wmma_f32_16x16x32_f16(false, a1, false, b1, (short)0, acc11, false, false);
  }

  // C/D layout: acc[r] -> row tileRow + r + (lane>=16 ? 8:0), col tileCol + (lane&15)
  int radd = hiLane << 3;
  v8f accs[2][2] = {{acc00, acc01}, {acc10, acc11}};
#pragma unroll
  for (int ni = 0; ni < 2; ++ni) {
    int col = c0 + ni * 16 + lane15;
    if (col < N) {
      float bv = bias[col];
#pragma unroll
      for (int mi = 0; mi < 2; ++mi) {
        int rbase = r0 + mi * 16 + radd;
#pragma unroll
        for (int r = 0; r < 8; ++r) {
          int row = rbase + r;
          float val = accs[mi][ni][r] + bv;
          if (mode == 2) {
            int nimg = row >> 12;
            size_t oidx = ((size_t)(nimg * CDIM + col)) * HWSZ + (row & (HWSZ - 1));
            outGate[oidx] = val * gateX[oidx];
          } else {
            outF[(size_t)row * N + col] = val;
            if (mode == 1) outH[(size_t)row * CDIM + col] = (_Float16)val;
          }
        }
      }
    }
  }
}

// ---------------------------------------------------------------------------
// depthwise 3x3 conv (zero pad) + bias -> LayerNorm over C -> exact GELU
// one block per pixel, 256 threads = 256 channels
// ---------------------------------------------------------------------------
__global__ __launch_bounds__(256) void dwconv_ln_gelu(const float* __restrict__ X,
                                                      const float* __restrict__ w,
                                                      const float* __restrict__ b,
                                                      const float* __restrict__ g,
                                                      const float* __restrict__ beta,
                                                      _Float16* __restrict__ Fh) {
  int l = blockIdx.x;
  int c = threadIdx.x;
  int n  = l >> 12;
  int hw = l & (HWSZ - 1);
  int py = hw >> 6, px = hw & 63;

  float s = 0.f;
#pragma unroll
  for (int ky = 0; ky < 3; ++ky) {
    int yy = py + ky - 1;
    if ((unsigned)yy >= 64u) continue;
#pragma unroll
    for (int kx = 0; kx < 3; ++kx) {
      int xx = px + kx - 1;
      if ((unsigned)xx >= 64u) continue;
      s += X[((size_t)((n << 12) + (yy << 6) + xx)) * CDIM + c] *
           w[(ky * 3 + kx) * CDIM + c];
    }
  }
  s += b[c];

  __shared__ float rs[256];
  __shared__ float rq[256];
  rs[c] = s; rq[c] = s * s;
  __syncthreads();
  for (int o = 128; o > 0; o >>= 1) {
    if (c < o) { rs[c] += rs[c + o]; rq[c] += rq[c + o]; }
    __syncthreads();
  }
  float mean = rs[0] * (1.f / 256.f);
  float var  = rq[0] * (1.f / 256.f) - mean * mean;
  float inv  = rsqrtf(var + 1e-5f);
  float v    = (s - mean) * inv * g[c] + beta[c];
  float ge   = 0.5f * v * (1.f + erff(v * 0.70710678118654752f));
  Fh[(size_t)l * CDIM + c] = (_Float16)ge;
}

// ---------------------------------------------------------------------------
// softmax over K=9 per group + bilinear deformable sampling (zero pad)
// one block per pixel, wave-per-group: 8 groups x 32 channels
// ---------------------------------------------------------------------------
__global__ __launch_bounds__(256) void dcn_sample(const float* __restrict__ V,
                                                  const float* __restrict__ OFF,
                                                  const float* __restrict__ MKL,
                                                  _Float16* __restrict__ Out) {
  int l  = blockIdx.x;
  int gg = threadIdx.x >> 5;
  int cc = threadIdx.x & 31;
  int n  = l >> 12;
  int hw = l & (HWSZ - 1);
  int py = hw >> 6, px = hw & 63;

  const float* ob = OFF + (size_t)l * 144 + gg * 18;
  const float* mb = MKL + (size_t)l * 72 + gg * 9;

  float m[9];
  float mx = -3.0e38f;
#pragma unroll
  for (int k = 0; k < 9; ++k) { m[k] = mb[k]; mx = fmaxf(mx, m[k]); }
  float sm = 0.f;
#pragma unroll
  for (int k = 0; k < 9; ++k) { m[k] = expf(m[k] - mx); sm += m[k]; }
  float minv = 1.f / sm;

  const float* vb = V + ((size_t)(n << 12)) * CDIM + gg * 32 + cc;
  float acc = 0.f;
#pragma unroll
  for (int k = 0; k < 9; ++k) {
    float ux = (float)px + (float)(k / 3 - 1) + ob[2 * k];
    float uy = (float)py + (float)(k % 3 - 1) + ob[2 * k + 1];
    float xf = floorf(ux), yf = floorf(uy);
    float tx = ux - xf, ty = uy - yf;
    int x0 = (int)xf, y0 = (int)yf;
    float sv = 0.f;
#pragma unroll
    for (int dy = 0; dy < 2; ++dy) {
      int yi = y0 + dy;
      if ((unsigned)yi >= 64u) continue;
      float wy = dy ? ty : (1.f - ty);
#pragma unroll
      for (int dx = 0; dx < 2; ++dx) {
        int xi = x0 + dx;
        if ((unsigned)xi >= 64u) continue;
        float wx = dx ? tx : (1.f - tx);
        sv += wx * wy * vb[((size_t)(yi * 64 + xi)) * CDIM];
      }
    }
    acc += m[k] * minv * sv;
  }
  Out[(size_t)l * CDIM + gg * 32 + cc] = (_Float16)acc;
}

// ---------------------------------------------------------------------------
extern "C" void kernel_launch(void* const* d_in, const int* in_sizes, int n_in,
                              void* d_out, int out_size, void* d_ws, size_t ws_size,
                              hipStream_t stream) {
  (void)in_sizes; (void)n_in; (void)out_size; (void)ws_size;
  const float* x = (const float*)d_in[0];
  auto P = [&](int i) { return (const float*)d_in[i]; };

  char* base = (char*)d_ws;
  size_t cur = 0;
  auto alloc = [&](size_t bytes) -> void* {
    void* p = base + cur;
    cur += (bytes + 255) & ~(size_t)255;
    return p;
  };

  float*     XF  = (float*)    alloc((size_t)LROWS * CDIM * 4);
  _Float16*  XH  = (_Float16*) alloc((size_t)LROWS * CDIM * 2);
  _Float16*  FH  = (_Float16*) alloc((size_t)LROWS * CDIM * 2);
  float*     VF  = (float*)    alloc((size_t)LROWS * CDIM * 4);
  float*     OFb = (float*)    alloc((size_t)LROWS * 144 * 4);
  float*     MKb = (float*)    alloc((size_t)LROWS * 72 * 4);
  _Float16*  SH  = (_Float16*) alloc((size_t)LROWS * CDIM * 2);

  // packed weight sizes in halves; NT padded to even so 2-wide N-tiling is safe
  const size_t PK16 = (size_t)16 * KT8 * 512;   // N=256 -> 16 tiles
  const size_t PK10 = (size_t)10 * KT8 * 512;   // N=144 -> 9 tiles, pad to 10
  const size_t PK6  = (size_t)6  * KT8 * 512;   // N=72  -> 5 tiles, pad to 6
  _Float16* PIN[2];  _Float16* POF[2];  _Float16* PMK[2];  _Float16* POU[2];
  for (int s = 0; s < 2; ++s) {
    PIN[s] = (_Float16*)alloc(PK16 * 2);
    POF[s] = (_Float16*)alloc(PK10 * 2);
    PMK[s] = (_Float16*)alloc(PK6  * 2);
    POU[s] = (_Float16*)alloc(PK16 * 2);
  }
  _Float16* PPJ = (_Float16*)alloc(PK16 * 2);

  // 1) transpose input
  nchw_to_nhwc<<<8192, 256, 0, stream>>>(x, XF, XH);

  // 2) pack all weights (zero-fill padded columns)
  for (int s = 0; s < 2; ++s) {
    int pb = 1 + s * 12;
    pack_w<<<(int)((PK16 + 255) / 256), 256, 0, stream>>>(P(pb + 4),  PIN[s], 256, 16);
    pack_w<<<(int)((PK10 + 255) / 256), 256, 0, stream>>>(P(pb + 6),  POF[s], 144, 10);
    pack_w<<<(int)((PK6  + 255) / 256), 256, 0, stream>>>(P(pb + 8),  PMK[s], 72,  6);
    pack_w<<<(int)((PK16 + 255) / 256), 256, 0, stream>>>(P(pb + 10), POU[s], 256, 16);
  }
  pack_w<<<(int)((PK16 + 255) / 256), 256, 0, stream>>>(P(25), PPJ, 256, 16);

  // GEMM grids: blocks = (row-pairs * n-tile-pairs) / 8 waves = 32 * NP
  const int G256 = 32 * 8;   // NP=8
  const int G144 = 32 * 5;   // NP=5 (tiles 9+pad)
  const int G72  = 32 * 3;   // NP=3 (tiles 5+pad)

  // 3) two DCNv3 stages
  for (int s = 0; s < 2; ++s) {
    int pb = 1 + s * 12;
    dwconv_ln_gelu<<<8192, 256, 0, stream>>>(XF, P(pb + 0), P(pb + 1), P(pb + 2), P(pb + 3), FH);
    gemm_wmma<<<G256, 256, 0, stream>>>(XH, PIN[s], P(pb + 5), 256, 0, VF,  nullptr, nullptr, nullptr);
    gemm_wmma<<<G144, 256, 0, stream>>>(FH, POF[s], P(pb + 7), 144, 0, OFb, nullptr, nullptr, nullptr);
    gemm_wmma<<<G72,  256, 0, stream>>>(FH, PMK[s], P(pb + 9), 72,  0, MKb, nullptr, nullptr, nullptr);
    dcn_sample<<<8192, 256, 0, stream>>>(VF, OFb, MKb, SH);
    gemm_wmma<<<G256, 256, 0, stream>>>(SH, POU[s], P(pb + 11), 256, 1, XF, XH, nullptr, nullptr);
  }

  // 4) final projection + gate, writes NCHW to d_out
  gemm_wmma<<<G256, 256, 0, stream>>>(XH, PPJ, P(26), 256, 2,
                                      nullptr, nullptr, x, (float*)d_out);
}
